// Submission_66340064854150
// MI455X (gfx1250) — matrix operand order repaired and checked by reading, not compiled, not run on hardware
//
#include <hip/hip_runtime.h>
#include <hip/hip_bf16.h>
#include <math.h>

// Harness owns main/IO/timing; all kernels on `stream`; d_ws scratch ~48.6 MB.

typedef __attribute__((ext_vector_type(16))) __bf16 v16bf;
typedef __attribute__((ext_vector_type(8)))  float  v8f;

union BF16Frag { v16bf v; uint4 q[2]; };

#define NN 8192
#define DD 1024
#define HH 512
#define KK 16
#define CC 9
#define BPAD 72   // padded d-stride (elements) for LDS B tiles: conflict-free frag reads

static __device__ __forceinline__ unsigned short f2bf(float f) {
  unsigned int u = __float_as_uint(f);
  unsigned int r = (u + 0x7FFFu + ((u >> 16) & 1u)) >> 16;   // RNE
  return (unsigned short)r;
}

// ---------------- 1) convert x (f32 -> bf16), 4 elems/thread ----------------
__global__ __launch_bounds__(256) void k_conv_x(const float* __restrict__ x,
                                                unsigned short* __restrict__ xb) {
  const int i = blockIdx.x * 256 + threadIdx.x;
  const float4 f = ((const float4*)x)[i];
  unsigned long long p =
      (unsigned long long)f2bf(f.x) |
      ((unsigned long long)f2bf(f.y) << 16) |
      ((unsigned long long)f2bf(f.z) << 32) |
      ((unsigned long long)f2bf(f.w) << 48);
  ((unsigned long long*)xb)[i] = p;
}

// ------- 2) transpose+convert weights: [k][d][h] f32 -> [k][h][d] bf16 -------
__global__ __launch_bounds__(256) void k_transpose_w(const float* __restrict__ Vw,
                                                     const float* __restrict__ Uw,
                                                     unsigned short* __restrict__ vwt,
                                                     unsigned short* __restrict__ uwt) {
  __shared__ float tile[32][33];
  const int h0 = blockIdx.x * 32, d0 = blockIdx.y * 32, z = blockIdx.z;
  const float* src; unsigned short* dst; int k;
  if (z < KK) { k = z;      src = Vw; dst = vwt; }
  else        { k = z - KK; src = Uw; dst = uwt; }
  src += (size_t)k * DD * HH;
  dst += (size_t)k * HH * DD;
  const int tx = threadIdx.x, ty = threadIdx.y;
  for (int r = ty; r < 32; r += 8)
    tile[r][tx] = src[(size_t)(d0 + r) * HH + h0 + tx];
  __syncthreads();
  for (int r = ty; r < 32; r += 8)
    dst[(size_t)(h0 + r) * DD + d0 + tx] = f2bf(tile[tx][r]);
}

// -------- async-stage one 64h x 64d bf16 tile pair (V,U) into LDS ----------
// Every thread copies 32 B per matrix: row = tid>>2, d-segment = (tid&3)*16.
// INST_OFFSET applies to BOTH the LDS and global address (ISA 15.18 pseudocode),
// so offset:16 gives the second b128 of the 32 B chunk on both sides.
static __device__ __forceinline__ void stage_tiles(const unsigned short* __restrict__ vwt,
                                                   const unsigned short* __restrict__ uwt,
                                                   int k, int h0, int d0,
                                                   int srow, int sseg,
                                                   unsigned ldsV, unsigned ldsU) {
  const unsigned long long gv =
      (unsigned long long)(vwt + ((size_t)k * HH + h0 + srow) * DD + d0 + sseg);
  const unsigned long long gu =
      (unsigned long long)(uwt + ((size_t)k * HH + h0 + srow) * DD + d0 + sseg);
  asm volatile("global_load_async_to_lds_b128 %0, %1, off"            :: "v"(ldsV), "v"(gv) : "memory");
  asm volatile("global_load_async_to_lds_b128 %0, %1, off offset:16"  :: "v"(ldsV), "v"(gv) : "memory");
  asm volatile("global_load_async_to_lds_b128 %0, %1, off"            :: "v"(ldsU), "v"(gu) : "memory");
  asm volatile("global_load_async_to_lds_b128 %0, %1, off offset:16"  :: "v"(ldsU), "v"(gu) : "memory");
}

// ---- 3) fused gated-attention scores: bf16 WMMA + tanh*sigmoid*ww reduce ----
// grid: (N/256, K); block 256 = 8 waves; wave owns 32 rows x 64 h, loops all H.
// B (weights) double-buffered in LDS via async copies, shared by all 8 waves.
__global__ __launch_bounds__(256) void k_scores(const unsigned short* __restrict__ xb,
                                                const unsigned short* __restrict__ vwt,
                                                const unsigned short* __restrict__ uwt,
                                                const float* __restrict__ Vb,
                                                const float* __restrict__ Ub,
                                                const float* __restrict__ ww,
                                                const float* __restrict__ wb,
                                                float* __restrict__ scores) {
  __shared__ __align__(16) unsigned short sB[2][2][64 * BPAD];  // [buf][mat][h*BPAD+d]

  const int k    = blockIdx.y;
  const int wave = threadIdx.x >> 5;
  const int lane = threadIdx.x & 31;
  const int l15  = lane & 15;
  const int nbase = blockIdx.x * 256 + wave * 32;
  const unsigned short* ap0 = xb + (size_t)(nbase + l15) * DD;
  const unsigned short* ap1 = ap0 + (size_t)16 * DD;
  const int ka = (lane < 16) ? 0 : 8;    // A-frag K chunks {ka, ka+16}
  const int kb = (lane < 16) ? 0 : 8;    // B-frag K chunks {kb, kb+16}

  // staging role (256 threads cover 64 rows x 64 d, 32 B each)
  const int srow = threadIdx.x >> 2;
  const int sseg = (threadIdx.x & 3) * 16;
  unsigned ldsV[2], ldsU[2];
#pragma unroll
  for (int b = 0; b < 2; ++b) {
    ldsV[b] = (unsigned)(size_t)&sB[b][0][srow * BPAD + sseg];
    ldsU[b] = (unsigned)(size_t)&sB[b][1][srow * BPAD + sseg];
  }

  float p[2][8];
#pragma unroll
  for (int rr = 0; rr < 2; ++rr)
#pragma unroll
    for (int r = 0; r < 8; ++r) p[rr][r] = 0.f;

  int buf = 0;
  stage_tiles(vwt, uwt, k, 0, 0, srow, sseg, ldsV[0], ldsU[0]);

  for (int h0 = 0; h0 < HH; h0 += 64) {
    v8f accv[2][4], accu[2][4];
    {
      v8f z = {};
#pragma unroll
      for (int rr = 0; rr < 2; ++rr)
#pragma unroll
        for (int t = 0; t < 4; ++t) { accv[rr][t] = z; accu[rr][t] = z; }
    }
    for (int d0 = 0; d0 < DD; d0 += 64) {
      asm volatile("s_wait_asynccnt 0" ::: "memory");
      __syncthreads();                       // tiles landed, prior reads done
      const int nbuf = buf ^ 1;
      int nd = d0 + 64, nh = h0;
      if (nd == DD) { nd = 0; nh = h0 + 64; }
      if (nh < HH)                            // prefetch next tile pair
        stage_tiles(vwt, uwt, k, nh, nd, srow, sseg, ldsV[nbuf], ldsU[nbuf]);

#pragma unroll
      for (int sub = 0; sub < 2; ++sub) {     // two K=32 sub-steps
        const int dk = d0 + sub * 32;
        BF16Frag a0, a1;
        a0.q[0] = *(const uint4*)(ap0 + dk + ka);
        a0.q[1] = *(const uint4*)(ap0 + dk + ka + 16);
        a1.q[0] = *(const uint4*)(ap1 + dk + ka);
        a1.q[1] = *(const uint4*)(ap1 + dk + ka + 16);
#pragma unroll
        for (int t = 0; t < 4; ++t) {
          const int boff = (t * 16 + l15) * BPAD + sub * 32 + kb;
          BF16Frag fv, fu;
          fv.q[0] = *(const uint4*)&sB[buf][0][boff];
          fv.q[1] = *(const uint4*)&sB[buf][0][boff + 16];
          fu.q[0] = *(const uint4*)&sB[buf][1][boff];
          fu.q[1] = *(const uint4*)&sB[buf][1][boff + 16];
          accv[0][t] = __builtin_amdgcn_wmma_f32_16x16x32_bf16(
              false, a0.v, false, fv.v, (short)0, accv[0][t], false, false);
          accv[1][t] = __builtin_amdgcn_wmma_f32_16x16x32_bf16(
              false, a1.v, false, fv.v, (short)0, accv[1][t], false, false);
          accu[0][t] = __builtin_amdgcn_wmma_f32_16x16x32_bf16(
              false, a0.v, false, fu.v, (short)0, accu[0][t], false, false);
          accu[1][t] = __builtin_amdgcn_wmma_f32_16x16x32_bf16(
              false, a1.v, false, fu.v, (short)0, accu[1][t], false, false);
        }
      }
      buf = nbuf;
    }
    // gate + weighted reduce over this 64-wide h chunk (registers only)
#pragma unroll
    for (int t = 0; t < 4; ++t) {
      const int h = h0 + t * 16 + l15;
      const float vb = Vb[k * HH + h];
      const float ub = Ub[k * HH + h];
      const float w  = ww[k * HH + h];
#pragma unroll
      for (int rr = 0; rr < 2; ++rr)
#pragma unroll
        for (int r = 0; r < 8; ++r) {
          const float av = accv[rr][t][r] + vb;
          const float v  = 1.0f - 2.0f / (__expf(2.0f * av) + 1.0f);   // tanh
          const float u  = 1.0f / (1.0f + __expf(-(accu[rr][t][r] + ub)));
          p[rr][r] += v * u * w;
        }
    }
  }
  // reduce the 16 h-columns held across lanes of each half-wave
#pragma unroll
  for (int m = 1; m < 16; m <<= 1)
#pragma unroll
    for (int rr = 0; rr < 2; ++rr)
#pragma unroll
      for (int r = 0; r < 8; ++r) p[rr][r] += __shfl_xor(p[rr][r], m, 32);
  if (l15 == 0) {
    const float b = wb[k];
    const int half = (lane >> 4) << 3;      // lanes 0..15 -> M=r; 16..31 -> M=r+8
#pragma unroll
    for (int rr = 0; rr < 2; ++rr)
#pragma unroll
      for (int r = 0; r < 8; ++r)
        scores[(size_t)k * NN + nbase + rr * 16 + half + r] = p[rr][r] + b;
  }
}

// ---------------- 4) softmax over N per head (in place) ----------------
__global__ __launch_bounds__(256) void k_softmax(float* __restrict__ s) {
  __shared__ float red[256];
  float* row = s + (size_t)blockIdx.x * NN;
  const int t = threadIdx.x;
  float mx = -3.4e38f;
  for (int n = t; n < NN; n += 256) mx = fmaxf(mx, row[n]);
  red[t] = mx; __syncthreads();
  for (int off = 128; off; off >>= 1) { if (t < off) red[t] = fmaxf(red[t], red[t + off]); __syncthreads(); }
  mx = red[0]; __syncthreads();
  float sum = 0.f;
  for (int n = t; n < NN; n += 256) { float e = __expf(row[n] - mx); row[n] = e; sum += e; }
  red[t] = sum; __syncthreads();
  for (int off = 128; off; off >>= 1) { if (t < off) red[t] += red[t + off]; __syncthreads(); }
  const float inv = 1.0f / red[0];
  for (int n = t; n < NN; n += 256) row[n] *= inv;
}

// ------ 5) agg: sum_k (A[k] @ x) and mean(x) over patches; 1 thread/col ------
__global__ __launch_bounds__(256) void k_agg(const float* __restrict__ x,
                                             const float* __restrict__ A,
                                             float* __restrict__ aggsum,
                                             float* __restrict__ mean) {
  __shared__ float sA[KK * 256];
  const int d = blockIdx.x * 256 + threadIdx.x;
  float acc[KK];
#pragma unroll
  for (int k = 0; k < KK; ++k) acc[k] = 0.f;
  float accm = 0.f;
  for (int n0 = 0; n0 < NN; n0 += 256) {
    __syncthreads();
    for (int j = threadIdx.x; j < KK * 256; j += 256)
      sA[j] = A[(size_t)(j >> 8) * NN + n0 + (j & 255)];
    __syncthreads();
    for (int nn = 0; nn < 256; ++nn) {
      const float xv = x[(size_t)(n0 + nn) * DD + d];
      accm += xv;
#pragma unroll
      for (int k = 0; k < KK; ++k) acc[k] += sA[k * 256 + nn] * xv;
    }
  }
  float hs = 0.f;
#pragma unroll
  for (int k = 0; k < KK; ++k) hs += acc[k];
  aggsum[d] = hs;
  mean[d]   = accm * (1.0f / NN);
}

// ---------- 6) pooling-head MLP + combine + classifier + softmax ----------
__global__ __launch_bounds__(256) void k_head(const float* __restrict__ aggsum,
                                              const float* __restrict__ mean,
                                              const float* pW1, const float* pb1,
                                              const float* pW2, const float* pb2,
                                              const float* cW1, const float* cb1,
                                              const float* cW2, const float* cb2,
                                              const float* cW3, const float* cb3,
                                              const float* cW4, const float* cb4,
                                              float* __restrict__ out) {
  __shared__ float sa[1024];
  __shared__ float sb[1024];
  __shared__ float sl[16];
  const int t = threadIdx.x;
  for (int j = t; j < 1024; j += 256) sa[j] = mean[j];
  __syncthreads();
  for (int j = t; j < 512; j += 256) {                 // h1 = relu(m@pW1+pb1)
    float acc = pb1[j];
    for (int i = 0; i < 1024; ++i) acc += sa[i] * pW1[(size_t)i * 512 + j];
    sb[j] = fmaxf(acc, 0.f);
  }
  __syncthreads();
  for (int j = t; j < 1024; j += 256) {                // aggregated = (sum+avg_agg)/17
    float acc = pb2[j];
    for (int i = 0; i < 512; ++i) acc += sb[i] * pW2[(size_t)i * 1024 + j];
    sa[j] = (aggsum[j] + acc) * (1.0f / (KK + 1));
  }
  __syncthreads();
  for (int j = t; j < 1024; j += 256) {                // c1
    float acc = cb1[j];
    for (int i = 0; i < 1024; ++i) acc += sa[i] * cW1[(size_t)i * 1024 + j];
    sb[j] = fmaxf(acc, 0.f);
  }
  __syncthreads();
  for (int j = t; j < 512; j += 256) {                 // c2
    float acc = cb2[j];
    for (int i = 0; i < 1024; ++i) acc += sb[i] * cW2[(size_t)i * 512 + j];
    sa[j] = fmaxf(acc, 0.f);
  }
  __syncthreads();
  {                                                    // c3 (256 cols, 1/thread)
    float acc = cb3[t];
    for (int i = 0; i < 512; ++i) acc += sa[i] * cW3[(size_t)i * 256 + t];
    sb[t] = fmaxf(acc, 0.f);
  }
  __syncthreads();
  if (t < CC) {                                        // c4 logits
    float acc = cb4[t];
    for (int i = 0; i < 256; ++i) acc += sb[i] * cW4[(size_t)i * CC + t];
    sl[t] = acc;
  }
  __syncthreads();
  if (t == 0) {                                        // softmax over 9
    float mx = sl[0];
    for (int j = 1; j < CC; ++j) mx = fmaxf(mx, sl[j]);
    float s = 0.f, e[CC];
    for (int j = 0; j < CC; ++j) { e[j] = __expf(sl[j] - mx); s += e[j]; }
    const float inv = 1.0f / s;
    for (int j = 0; j < CC; ++j) out[j] = e[j] * inv;
  }
}

extern "C" void kernel_launch(void* const* d_in, const int* in_sizes, int n_in,
                              void* d_out, int out_size, void* d_ws, size_t ws_size,
                              hipStream_t stream) {
  const float* x   = (const float*)d_in[0];
  const float* Vw  = (const float*)d_in[1];
  const float* Vb  = (const float*)d_in[2];
  const float* Uw  = (const float*)d_in[3];
  const float* Ub  = (const float*)d_in[4];
  const float* ww  = (const float*)d_in[5];
  const float* wb  = (const float*)d_in[6];
  const float* pW1 = (const float*)d_in[7];
  const float* pb1 = (const float*)d_in[8];
  const float* pW2 = (const float*)d_in[9];
  const float* pb2 = (const float*)d_in[10];
  const float* cW1 = (const float*)d_in[11];
  const float* cb1 = (const float*)d_in[12];
  const float* cW2 = (const float*)d_in[13];
  const float* cb2 = (const float*)d_in[14];
  const float* cW3 = (const float*)d_in[15];
  const float* cb3 = (const float*)d_in[16];
  const float* cW4 = (const float*)d_in[17];
  const float* cb4 = (const float*)d_in[18];
  float* out = (float*)d_out;

  // workspace layout (~48.6 MB)
  char* ws = (char*)d_ws;
  size_t off = 0;
  unsigned short* xb  = (unsigned short*)(ws + off); off += (size_t)NN * DD * 2;       // 16 MB
  unsigned short* vwt = (unsigned short*)(ws + off); off += (size_t)KK * HH * DD * 2;  // 16 MB
  unsigned short* uwt = (unsigned short*)(ws + off); off += (size_t)KK * HH * DD * 2;  // 16 MB
  float* scores = (float*)(ws + off); off += (size_t)KK * NN * 4;                      // 512 KB
  float* aggsum = (float*)(ws + off); off += (size_t)DD * 4;
  float* meanv  = (float*)(ws + off); off += (size_t)DD * 4;
  (void)ws_size; (void)in_sizes; (void)n_in; (void)out_size;

  k_conv_x<<<(NN * DD / 4) / 256, 256, 0, stream>>>(x, xb);
  k_transpose_w<<<dim3(HH / 32, DD / 32, 2 * KK), dim3(32, 8), 0, stream>>>(Vw, Uw, vwt, uwt);
  k_scores<<<dim3(NN / 256, KK), 256, 0, stream>>>(xb, vwt, uwt, Vb, Ub, ww, wb, scores);
  k_softmax<<<KK, 256, 0, stream>>>(scores);
  k_agg<<<DD / 256, 256, 0, stream>>>(x, scores, aggsum, meanv);
  k_head<<<1, 256, 0, stream>>>(aggsum, meanv, pW1, pb1, pW2, pb2,
                                cW1, cb1, cW2, cb2, cW3, cb3, cW4, cb4, out);
}
